// TriangleAttentionStartingNode_7215545057297
// MI455X (gfx1250) — matrix operand order repaired and checked by reading, not compiled, not run on hardware
//
#include <hip/hip_runtime.h>
#include <hip/hip_bf16.h>
#include <math.h>
#include <stdint.h>

typedef __attribute__((ext_vector_type(16))) _Float16 v16h;
typedef __attribute__((ext_vector_type(8)))  float    v8f;

#define NRES  256
#define CDIM  128
#define NHEAD 4
#define ACH   32
#define NN    (NRES * NRES)                 // 65536
#define FACTOR 0.17677669529663687f        // 1/sqrt(32)

// ---------------- WMMA fragment loaders (contiguous k = 16*hi + s, same order in A and B) -------

// A fragment: 16x32 f16 tile, row-major source, element (m,k) = src[m*ld + k]
static __device__ __forceinline__ v16h load_a_f16(const _Float16* src, int ld, int lane) {
    const int m  = lane & 15;
    const int hi = lane >> 4;
    v16h a;
#pragma unroll
    for (int j = 0; j < 8; ++j) {
        const int kb = hi * 16 + j * 2;
        a[2 * j]     = src[m * ld + kb];
        a[2 * j + 1] = src[m * ld + kb + 1];
    }
    return a;
}

// B fragment = transpose of a 16x32 f16 tile: B(k,n) = src[n*ld + k]
// Per-lane addresses are 16 contiguous halves -> 2x b128 loads.
static __device__ __forceinline__ v16h load_bT_f16(const _Float16* src, int ld, int lane) {
    const int n  = lane & 15;
    const int hi = lane >> 4;
    v16h b;
#pragma unroll
    for (int j = 0; j < 8; ++j) {
        const int k = hi * 16 + j * 2;
        b[2 * j]     = src[n * ld + k];
        b[2 * j + 1] = src[n * ld + k + 1];
    }
    return b;
}

// B fragment from pre-packed weights: one contiguous 32B load per lane
static __device__ __forceinline__ v16h load_b_packed(const _Float16* PW, int kt, int c, int lane) {
    return *(const v16h*)(PW + (size_t)(((kt * 8 + c) * 32) + lane) * 16);
}

// ---------------- Kernel 0: pack a 128x128 f32 weight into B-fragment f16 ----
__global__ void pack_w_kernel(const float* __restrict__ W, _Float16* __restrict__ PW) {
    const int lane = threadIdx.x;          // 0..31
    const int kt   = blockIdx.x >> 3;      // 0..3
    const int c    = blockIdx.x & 7;       // 0..7
    const int n = lane & 15, hi = lane >> 4;
    _Float16* dst = PW + (size_t)(((kt * 8 + c) * 32) + lane) * 16;
#pragma unroll
    for (int e = 0; e < 16; ++e) {
        const int k = hi * 16 + e;
        dst[e] = (_Float16)W[(kt * 32 + k) * CDIM + c * 16 + n];
    }
}

// ---------------- Kernel 1: LayerNorm + pair-bias nb (shuffle reductions) ----

__global__ void ln_nb_kernel(const float* __restrict__ act,
                             const float* __restrict__ ln_g,
                             const float* __restrict__ ln_b,
                             const float* __restrict__ w2d,
                             _Float16*   __restrict__ aH,
                             float*      __restrict__ nb) {
    const int row  = blockIdx.x;    // = i*256 + j
    const int t    = threadIdx.x;   // channel 0..127
    const int lane = t & 31;
    const int w    = t >> 5;        // wave 0..3
    __shared__ float xw[4][2];
    __shared__ float nbw[4][NHEAD];

    const float v = act[(size_t)row * CDIM + t];
    float s1 = v, s2 = v * v;
#pragma unroll
    for (int off = 16; off > 0; off >>= 1) {
        s1 += __shfl_xor(s1, off, 32);
        s2 += __shfl_xor(s2, off, 32);
    }
    if (lane == 0) { xw[w][0] = s1; xw[w][1] = s2; }
    __syncthreads();
    const float sum1 = xw[0][0] + xw[1][0] + xw[2][0] + xw[3][0];
    const float sum2 = xw[0][1] + xw[1][1] + xw[2][1] + xw[3][1];
    const float mu  = sum1 * (1.0f / CDIM);
    const float var = sum2 * (1.0f / CDIM) - mu * mu;

    const float a = (v - mu) * rsqrtf(var + 1e-5f) * ln_g[t] + ln_b[t];
    aH[(size_t)row * CDIM + t] = (_Float16)a;

    float p[NHEAD];
#pragma unroll
    for (int h = 0; h < NHEAD; ++h) p[h] = a * w2d[t * NHEAD + h];
#pragma unroll
    for (int off = 16; off > 0; off >>= 1) {
#pragma unroll
        for (int h = 0; h < NHEAD; ++h) p[h] += __shfl_xor(p[h], off, 32);
    }
    if (lane == 0) {
#pragma unroll
        for (int h = 0; h < NHEAD; ++h) nbw[w][h] = p[h];
    }
    __syncthreads();
    if (t < NHEAD) nb[t * NN + row] = nbw[0][t] + nbw[1][t] + nbw[2][t] + nbw[3][t];
}

// ---------------- Kernel 2: fused q/k/v/g projections ------------------------
// which==2 (v) is stored TRANSPOSED: vT[((m*NHEAD+h)*ACH + c)*NRES + j]

__global__ void proj_kernel(const _Float16* __restrict__ aH,
                            const _Float16* __restrict__ pwq,
                            const _Float16* __restrict__ pwk,
                            const _Float16* __restrict__ pwv,
                            const _Float16* __restrict__ pwg,
                            const float* __restrict__ bg,
                            _Float16* __restrict__ qH,
                            _Float16* __restrict__ kH,
                            _Float16* __restrict__ vT,
                            _Float16* __restrict__ gH) {
    const int lane  = threadIdx.x & 31;
    const int wave  = threadIdx.x >> 5;     // 0..7 -> 16-col tile
    const int row0  = blockIdx.x * 16;
    const int col0  = wave * 16;
    const int which = blockIdx.y;           // 0=q 1=k 2=v 3=g

    const _Float16* PW = (which == 0) ? pwq : (which == 1) ? pwk : (which == 2) ? pwv : pwg;

    __builtin_prefetch(aH + (size_t)(row0 + 16) * CDIM, 0, 1);

    v8f acc = {};
#pragma unroll
    for (int kt = 0; kt < 4; ++kt) {
        v16h a = load_a_f16(aH + (size_t)row0 * CDIM + kt * 32, CDIM, lane);
        v16h b = load_b_packed(PW, kt, wave, lane);
        acc = __builtin_amdgcn_wmma_f32_16x16x32_f16(false, a, false, b, (short)0, acc, false, false);
    }

    const int n = lane & 15, hi = lane >> 4;
#pragma unroll
    for (int r = 0; r < 8; ++r) {
        const int row = row0 + hi * 8 + r;   // global row = m*256 + j
        const int col = col0 + n;            // h*32 + c
        float v = acc[r];
        if (which == 0) {
            v *= FACTOR;
            qH[(size_t)row * CDIM + col] = (_Float16)v;
        } else if (which == 1) {
            kH[(size_t)row * CDIM + col] = (_Float16)v;
        } else if (which == 2) {
            const int mm = row >> 8, jj = row & 255;
            const int hh = col >> 5, cc = col & 31;
            vT[((size_t)(mm * NHEAD + hh) * ACH + cc) * NRES + jj] = (_Float16)v;
        } else {
            v = __builtin_amdgcn_rcpf(1.0f + __expf(-(v + bg[col])));   // sigmoid gate
            gH[(size_t)row * CDIM + col] = (_Float16)v;
        }
    }
}

// ---------------- Kernel 3: attention --------------------------------------
// block = 64 threads (2 waves). K (row-major) and V (transposed) slabs for
// (m,h) staged into LDS once per block via global_load_async_to_lds_b128;
// each wave then processes 4 query tiles of 16 rows against the staged slabs.
// Scores stay in registers; softmax uses v_exp_f32 + shfl reductions; P round-
// trips through LDS only for the D->A fragment re-layout.

__global__ void __launch_bounds__(64) attn_kernel(
                            const _Float16* __restrict__ qH,
                            const _Float16* __restrict__ kH,
                            const _Float16* __restrict__ vT,
                            const _Float16* __restrict__ gH,
                            const float* __restrict__ nb,
                            const float* __restrict__ mask,
                            _Float16* __restrict__ xH) {
    const int lane = threadIdx.x & 31;
    const int wave = threadIdx.x >> 5;                 // 0..1
    const int m    = blockIdx.z;
    const int h    = blockIdx.y;

    __shared__ _Float16 Kt[NRES * ACH];    // 16 KB, row-major [j][c]
    __shared__ _Float16 Vt[ACH * NRES];    // 16 KB, transposed [c][j]
    __shared__ _Float16 P[2][16][NRES];    // 16 KB

    const _Float16* qbase = qH + (size_t)(m * NRES) * CDIM + h * ACH;
    const _Float16* kbase = kH + (size_t)(m * NRES) * CDIM + h * ACH;
    const _Float16* vtb   = vT + (size_t)(m * NHEAD + h) * ACH * NRES;  // contiguous 16 KB

    // ---- async stage K (strided rows) and V (linear slab) into LDS ----
    for (int it = 0; it < 16; ++it) {
        const int e   = threadIdx.x + it * 64;   // chunk 0..1023 (16B each)
        const int row = e >> 2;
        const int seg = e & 3;
        const uint64_t gk = (uint64_t)(uintptr_t)(kbase + row * CDIM + seg * 8);
        const uint64_t gv = (uint64_t)(uintptr_t)(vtb + e * 8);
        const uint32_t lk = (uint32_t)(uintptr_t)(&Kt[row * ACH + seg * 8]);
        const uint32_t lv = (uint32_t)(uintptr_t)(&Vt[e * 8]);
        asm volatile("global_load_async_to_lds_b128 %0, %1, off" :: "v"(lk), "v"(gk) : "memory");
        asm volatile("global_load_async_to_lds_b128 %0, %1, off" :: "v"(lv), "v"(gv) : "memory");
    }
    asm volatile("s_wait_asynccnt 0x0" ::: "memory");
    __syncthreads();

    const int n = lane & 15, hi = lane >> 4;
    const float* nbh = nb + h * NN;

    for (int qi = 0; qi < 4; ++qi) {
        const int q0 = (blockIdx.x * 2 + wave) * 64 + qi * 16;   // query tile start

        const v16h aq = load_a_f16(qbase + q0 * CDIM, CDIM, lane);  // factor pre-folded

        // ---- S = Q K^T (registers) ----
        v8f acc[16];
#pragma unroll
        for (int jt = 0; jt < 16; ++jt) {
            v16h bk = load_bT_f16(&Kt[(jt * 16) * ACH], ACH, lane);
            v8f z = {};
            acc[jt] = __builtin_amdgcn_wmma_f32_16x16x32_f16(false, aq, false, bk, (short)0, z, false, false);
        }

        // ---- add mask bias + pair bias ----
#pragma unroll
        for (int jt = 0; jt < 16; ++jt) {
            const int j = jt * 16 + n;
            const float mb = 1e9f * (mask[m * NRES + j] - 1.0f);
#pragma unroll
            for (int r = 0; r < 8; ++r) {
                const int rr = hi * 8 + r;
                acc[jt][r] += mb + nbh[(q0 + rr) * NRES + j];
            }
        }

        // ---- softmax in registers; rows live in 16-lane groups ----
#pragma unroll
        for (int r = 0; r < 8; ++r) {
            float mx = -3.0e38f;
#pragma unroll
            for (int jt = 0; jt < 16; ++jt) mx = fmaxf(mx, acc[jt][r]);
            for (int off = 8; off > 0; off >>= 1) mx = fmaxf(mx, __shfl_xor(mx, off, 32));
            float sum = 0.0f;
#pragma unroll
            for (int jt = 0; jt < 16; ++jt) { const float e = __expf(acc[jt][r] - mx); acc[jt][r] = e; sum += e; }
            for (int off = 8; off > 0; off >>= 1) sum += __shfl_xor(sum, off, 32);
            const float inv = __builtin_amdgcn_rcpf(sum);
            const int rr = hi * 8 + r;
#pragma unroll
            for (int jt = 0; jt < 16; ++jt) P[wave][rr][jt * 16 + n] = (_Float16)(acc[jt][r] * inv);
        }
        // P[wave] is wave-private; in-order LDS within a wave needs no barrier.

        // ---- O = P * V (two 16x16 tiles); all fragments contiguous in LDS ----
        v8f o0 = {}, o1 = {};
#pragma unroll
        for (int kt = 0; kt < 8; ++kt) {
            v16h ap = load_a_f16(&P[wave][0][kt * 32], NRES, lane);
            v16h b0 = load_bT_f16(&Vt[kt * 32],             NRES, lane);   // cols 0..15
            v16h b1 = load_bT_f16(&Vt[16 * NRES + kt * 32], NRES, lane);   // cols 16..31
            o0 = __builtin_amdgcn_wmma_f32_16x16x32_f16(false, ap, false, b0, (short)0, o0, false, false);
            o1 = __builtin_amdgcn_wmma_f32_16x16x32_f16(false, ap, false, b1, (short)0, o1, false, false);
        }

        // ---- gate and store X ----
#pragma unroll
        for (int r = 0; r < 8; ++r) {
            const int rr = hi * 8 + r;
            const size_t base = (size_t)(m * NRES + q0 + rr) * CDIM + h * ACH;
            const float g0 = (float)gH[base + n];
            const float g1 = (float)gH[base + 16 + n];
            xH[base + n]      = (_Float16)(o0[r] * g0);
            xH[base + 16 + n] = (_Float16)(o1[r] * g1);
        }
    }
}

// ---------------- Kernel 4: output projection --------------------------------

__global__ void out_kernel(const _Float16* __restrict__ xH,
                           const _Float16* __restrict__ pwo,
                           const float* __restrict__ bo,
                           float* __restrict__ out) {
    const int lane = threadIdx.x & 31;
    const int wave = threadIdx.x >> 5;
    const int row0 = blockIdx.x * 16;
    const int col0 = wave * 16;

    __builtin_prefetch(xH + (size_t)(row0 + 16) * CDIM, 0, 1);

    v8f acc = {};
#pragma unroll
    for (int kt = 0; kt < 4; ++kt) {
        v16h a = load_a_f16(xH + (size_t)row0 * CDIM + kt * 32, CDIM, lane);
        v16h b = load_b_packed(pwo, kt, wave, lane);
        acc = __builtin_amdgcn_wmma_f32_16x16x32_f16(false, a, false, b, (short)0, acc, false, false);
    }
    const int n = lane & 15, hi = lane >> 4;
#pragma unroll
    for (int r = 0; r < 8; ++r) {
        const int row = row0 + hi * 8 + r;
        const int col = col0 + n;
        out[(size_t)row * CDIM + col] = acc[r] + bo[col];
    }
}

// ---------------- launcher ---------------------------------------------------

extern "C" void kernel_launch(void* const* d_in, const int* in_sizes, int n_in,
                              void* d_out, int out_size, void* d_ws, size_t ws_size,
                              hipStream_t stream) {
    const float* act  = (const float*)d_in[0];
    const float* mask = (const float*)d_in[1];
    const float* ln_g = (const float*)d_in[2];
    const float* ln_b = (const float*)d_in[3];
    const float* wq   = (const float*)d_in[4];
    const float* wk   = (const float*)d_in[5];
    const float* wv   = (const float*)d_in[6];
    const float* w2d  = (const float*)d_in[7];
    const float* wg   = (const float*)d_in[8];
    const float* bg   = (const float*)d_in[9];
    const float* wo   = (const float*)d_in[10];
    const float* bo   = (const float*)d_in[11];
    float* out = (float*)d_out;

    const size_t NNC = (size_t)NN * CDIM;
    const size_t WPK = 4 * 8 * 32 * 16;   // 16384 halves per packed weight
    _Float16* aH  = (_Float16*)d_ws;      // 16 MB; dead after proj -> reused as X
    _Float16* qH  = aH + NNC;
    _Float16* kH  = qH + NNC;
    _Float16* vT  = kH + NNC;             // transposed V
    _Float16* gH  = vT + NNC;
    float*    nb  = (float*)(gH + NNC);   // 4*NN f32 = 1 MB
    _Float16* pwq = (_Float16*)(nb + 4 * NN);
    _Float16* pwk = pwq + WPK;
    _Float16* pwv = pwk + WPK;
    _Float16* pwg = pwv + WPK;
    _Float16* pwo = pwg + WPK;
    _Float16* xH  = aH;                   // alias: gated attention output

    pack_w_kernel<<<32, 32, 0, stream>>>(wq, pwq);
    pack_w_kernel<<<32, 32, 0, stream>>>(wk, pwk);
    pack_w_kernel<<<32, 32, 0, stream>>>(wv, pwv);
    pack_w_kernel<<<32, 32, 0, stream>>>(wg, pwg);
    pack_w_kernel<<<32, 32, 0, stream>>>(wo, pwo);

    ln_nb_kernel<<<NN, CDIM, 0, stream>>>(act, ln_g, ln_b, w2d, aH, nb);
    proj_kernel <<<dim3(NN / 16, 4), 256, 0, stream>>>(aH, pwq, pwk, pwv, pwg, bg, qH, kH, vT, gH);
    attn_kernel <<<dim3(2, NHEAD, NRES), 64, 0, stream>>>(qH, kH, vT, gH, nb, mask, xH);
    out_kernel  <<<dim3(NN / 16), 256, 0, stream>>>(xH, pwo, bo, out);
}
